// AttentionBlock_1692217114773
// MI455X (gfx1250) — matrix operand order repaired and checked by reading, not compiled, not run on hardware
//
#include <hip/hip_runtime.h>
#include <hip/hip_bf16.h>

typedef __attribute__((ext_vector_type(16))) __bf16 v16bf;
typedef __attribute__((ext_vector_type(8)))  float  v8f;
typedef __attribute__((ext_vector_type(4)))  unsigned int v4u;
typedef __attribute__((ext_vector_type(8)))  int v8i;
typedef __attribute__((ext_vector_type(4)))  int v4i;

#define B_   8
#define C_   512
#define T_   1024
#define NH_  8
#define CH_  64

#if defined(__has_builtin)
#if __has_builtin(__builtin_amdgcn_tensor_load_to_lds)
#define HAVE_TDM 1
#endif
#endif
#ifndef HAVE_TDM
#define HAVE_TDM 0
#endif

// ---------------------------------------------------------------------------
// CDNA5 async global->LDS copy (ASYNCcnt-tracked, no VGPR destination).
// lds_off is the byte offset within the workgroup LDS window (low 32 bits of
// the generic shared-memory address, per the aperture mapping rules).
// ---------------------------------------------------------------------------
static __device__ __forceinline__ void async_load_b128(const void* gptr, void* lptr) {
    unsigned int       lds_off = (unsigned int)(unsigned long long)(uintptr_t)lptr;
    unsigned long long gaddr   = (unsigned long long)(uintptr_t)gptr;
    asm volatile("global_load_async_to_lds_b128 %0, %1, off"
                 :: "v"(lds_off), "v"(gaddr)
                 : "memory");
}
static __device__ __forceinline__ void wait_async0() {
    asm volatile("s_wait_asynccnt 0x0" ::: "memory");
}

#if HAVE_TDM
// ---------------------------------------------------------------------------
// Tensor Data Mover: DMA a 2D tile (tl1 rows x tl0 elements, 2-byte elems,
// row stride stride0) from global memory into LDS (packed row-major).
// Descriptor per CDNA5 ISA ch.8: group0 {count=1, lds_addr, global_addr,
// type=2}; group1 {data_size=1(2B), tensor dims, tile dims, dim0 stride}.
// Issued by one wave; tracked with TENSORcnt.
// ---------------------------------------------------------------------------
static __device__ __forceinline__ void tdm_load_tile_2d(void* lptr, const void* gptr,
                                                        unsigned int td0, unsigned int td1,
                                                        unsigned int tl0, unsigned int tl1,
                                                        unsigned int stride0) {
    unsigned long long ga = (unsigned long long)(uintptr_t)gptr;
    v4u g0 = { 1u,                                             // count=1 (user D#)
               (unsigned int)(uintptr_t)lptr,                  // lds_addr
               (unsigned int)ga,                               // global_addr[31:0]
               (((unsigned int)(ga >> 32)) & 0x01FFFFFFu) | (2u << 30) }; // [56:32] | type=2
    v8i g1 = { (int)(1u << 16),                                // data_size=1 -> 2 bytes
               (int)((td0 & 0xFFFFu) << 16),                   // tensor_dim0[15:0]
               (int)(((td0 >> 16) & 0xFFFFu) | ((td1 & 0xFFFFu) << 16)),
               (int)(((td1 >> 16) & 0xFFFFu) | ((tl0 & 0xFFFFu) << 16)), // tile_dim0
               (int)(tl1 & 0xFFFFu),                           // tile_dim1 (tile_dim2=0)
               (int)stride0,                                   // tensor_dim0_stride[31:0]
               0, 0 };
    v4i z4 = { 0, 0, 0, 0 };
#if __clang_major__ >= 23
    v8i z8 = { 0, 0, 0, 0, 0, 0, 0, 0 };
    __builtin_amdgcn_tensor_load_to_lds(g0, g1, z4, z4, z8, 0);
#else
    __builtin_amdgcn_tensor_load_to_lds(g0, g1, z4, z4, 0);
#endif
}
#endif

// ---------------------------------------------------------------------------
// fp32 -> bf16 conversion (weights)
// ---------------------------------------------------------------------------
__global__ void __launch_bounds__(256) cvt_bf16_kernel(const float* __restrict__ src,
                                                       __bf16* __restrict__ dst, int n) {
    int i = blockIdx.x * 256 + threadIdx.x;
    if (i < n) dst[i] = (__bf16)src[i];
}

// ---------------------------------------------------------------------------
// GroupNorm: one block per (batch, group). 32 groups, 16 ch/group, T=1024.
// ---------------------------------------------------------------------------
__global__ void __launch_bounds__(256) groupnorm_kernel(const float* __restrict__ x,
                                                        const float* __restrict__ gw,
                                                        const float* __restrict__ gb,
                                                        __bf16* __restrict__ xn) {
    int bg = blockIdx.x;
    int b = bg >> 5, g = bg & 31;
    const int CPG = C_ / 32;
    const int N   = CPG * T_;
    const float* xp = x + ((size_t)b * C_ + g * CPG) * T_;

    float s = 0.f, ss = 0.f;
    for (int i = threadIdx.x; i < N; i += 256) {
        float v = xp[i];
        s += v; ss += v * v;
    }
    __shared__ float r1[256], r2[256];
    r1[threadIdx.x] = s; r2[threadIdx.x] = ss;
    __syncthreads();
    for (int off = 128; off > 0; off >>= 1) {
        if (threadIdx.x < off) {
            r1[threadIdx.x] += r1[threadIdx.x + off];
            r2[threadIdx.x] += r2[threadIdx.x + off];
        }
        __syncthreads();
    }
    float mu   = r1[0] / (float)N;
    float var  = r2[0] / (float)N - mu * mu;
    float rinv = rsqrtf(var + 1e-5f);

    __bf16* op = xn + ((size_t)b * C_ + g * CPG) * T_;
    for (int i = threadIdx.x; i < N; i += 256) {
        int c = g * CPG + (i >> 10);
        float v = (xp[i] - mu) * rinv * gw[c] + gb[c];
        op[i] = (__bf16)v;
    }
}

// ---------------------------------------------------------------------------
// QKV GEMM: qkv[b][m][n] = sum_k W[m][k] * xn[b][k][n] + bias[m]  (bf16 out)
// M=1536, K=512, N=1024. Block tile 64x64, K-step 32, 8 waves (4Mx2N).
// Tiles staged with the Tensor Data Mover (wave 0 issues two 2D-tile DMAs,
// TENSORcnt-tracked); falls back to async-to-LDS when TDM is unavailable.
// ---------------------------------------------------------------------------
__global__ void __launch_bounds__(256) qkv_gemm_kernel(const __bf16* __restrict__ Aw,
                                                       const __bf16* __restrict__ Bx,
                                                       const float*  __restrict__ bias,
                                                       __bf16* __restrict__ out) {
    const int M = 3 * C_, K = C_, N = T_;
    int bb = blockIdx.z;
    int mb = blockIdx.y * 64, nb = blockIdx.x * 64;

    __shared__ __align__(16) __bf16 sA[64 * 32];
    __shared__ __align__(16) __bf16 sB[32 * 64];

    const __bf16* Bp = Bx + (size_t)bb * K * N;
    int tid = threadIdx.x;
    int w = tid >> 5, l = tid & 31, g = l >> 4, ln = l & 15;
    int wm = w >> 1, wn = w & 1;

#if !HAVE_TDM
    int ra = tid >> 2, sega = (tid & 3) * 8;
    int rb = tid >> 3, segb = (tid & 7) * 8;
    const __bf16* gA = &Aw[(size_t)(mb + ra) * K + sega];
    const __bf16* gB = &Bp[(size_t)rb * N + nb + segb];
    __bf16* lA = &sA[ra * 32 + sega];
    __bf16* lB = &sB[rb * 64 + segb];
#endif

    v8f acc0 = {}, acc1 = {};
    for (int kb = 0; kb < K; kb += 32) {
#if HAVE_TDM
        if (w == 0) {
            // A tile: 64 rows x 32 cols from W[mb..][kb..], row stride K
            tdm_load_tile_2d(sA, Aw + (size_t)mb * K + kb,
                             (unsigned)K, (unsigned)M, 32u, 64u, (unsigned)K);
            // B tile: 32 rows x 64 cols from X[kb..][nb..], row stride N
            tdm_load_tile_2d(sB, Bp + (size_t)kb * N + nb,
                             (unsigned)N, (unsigned)K, 64u, 32u, (unsigned)N);
            __builtin_amdgcn_s_wait_tensorcnt(0);
        }
#else
        async_load_b128(gA + kb, lA);
        async_load_b128(gB + (size_t)kb * N, lB);
        wait_async0();
#endif
        __syncthreads();

        v16bf a;
        const __bf16* ap = &sA[(wm * 16 + ln) * 32];
        #pragma unroll
        for (int e = 0; e < 16; ++e) a[e] = ap[(e < 8 ? e : e + 8) + g * 8];

        int kl = ln + g * 16;
        v16bf b0, b1;
        const __bf16* bp = &sB[wn * 32 + ln];
        #pragma unroll
        for (int e = 0; e < 16; ++e) { int k = (e < 8 ? e : e + 8) + g * 8; b0[e] = bp[k * 64]; b1[e] = bp[k * 64 + 16]; }

        acc0 = __builtin_amdgcn_wmma_f32_16x16x32_bf16(false, a, false, b0, (short)0, acc0, false, false);
        acc1 = __builtin_amdgcn_wmma_f32_16x16x32_bf16(false, a, false, b1, (short)0, acc1, false, false);
        __syncthreads();
    }
    #pragma unroll
    for (int j = 0; j < 8; ++j) {
        int m  = mb + wm * 16 + g * 8 + j;
        int n0 = nb + wn * 32 + ln;
        float bv = bias[m];
        out[((size_t)bb * M + m) * N + n0]      = (__bf16)(acc0[j] + bv);
        out[((size_t)bb * M + m) * N + n0 + 16] = (__bf16)(acc1[j] + bv);
    }
}

// ---------------------------------------------------------------------------
// Proj GEMM + bias + residual: out[b][m][n] = W[m][k]*a[b][k][n] + pb[m] + x
// M=K=512, N=1024. Tiles staged with global_load_async_to_lds_b128.
// ---------------------------------------------------------------------------
__global__ void __launch_bounds__(256) proj_gemm_kernel(const __bf16* __restrict__ Aw,
                                                        const __bf16* __restrict__ Bx,
                                                        const float*  __restrict__ bias,
                                                        const float*  __restrict__ xres,
                                                        float* __restrict__ out) {
    const int M = C_, K = C_, N = T_;
    int bb = blockIdx.z;
    int mb = blockIdx.y * 64, nb = blockIdx.x * 64;

    __shared__ __align__(16) __bf16 sA[64 * 32];
    __shared__ __align__(16) __bf16 sB[32 * 64];

    const __bf16* Bp = Bx + (size_t)bb * K * N;
    int tid = threadIdx.x;
    int w = tid >> 5, l = tid & 31, g = l >> 4, ln = l & 15;
    int wm = w >> 1, wn = w & 1;

    int ra = tid >> 2, sega = (tid & 3) * 8;
    int rb = tid >> 3, segb = (tid & 7) * 8;
    const __bf16* gA = &Aw[(size_t)(mb + ra) * K + sega];
    const __bf16* gB = &Bp[(size_t)rb * N + nb + segb];
    __bf16* lA = &sA[ra * 32 + sega];
    __bf16* lB = &sB[rb * 64 + segb];

    v8f acc0 = {}, acc1 = {};
    for (int kb = 0; kb < K; kb += 32) {
        async_load_b128(gA + kb, lA);
        async_load_b128(gB + (size_t)kb * N, lB);
        if (kb + 32 < K) {
            __builtin_prefetch(gA + kb + 32, 0, 1);
            __builtin_prefetch(gB + (size_t)(kb + 32) * N, 0, 1);
        }
        wait_async0();
        __syncthreads();

        v16bf a;
        const __bf16* ap = &sA[(wm * 16 + ln) * 32];
        #pragma unroll
        for (int e = 0; e < 16; ++e) a[e] = ap[(e < 8 ? e : e + 8) + g * 8];

        int kl = ln + g * 16;
        v16bf b0, b1;
        const __bf16* bp = &sB[wn * 32 + ln];
        #pragma unroll
        for (int e = 0; e < 16; ++e) { int k = (e < 8 ? e : e + 8) + g * 8; b0[e] = bp[k * 64]; b1[e] = bp[k * 64 + 16]; }

        acc0 = __builtin_amdgcn_wmma_f32_16x16x32_bf16(false, a, false, b0, (short)0, acc0, false, false);
        acc1 = __builtin_amdgcn_wmma_f32_16x16x32_bf16(false, a, false, b1, (short)0, acc1, false, false);
        __syncthreads();
    }
    #pragma unroll
    for (int j = 0; j < 8; ++j) {
        int m  = mb + wm * 16 + g * 8 + j;
        int n0 = nb + wn * 32 + ln;
        float bv = bias[m];
        size_t i0 = ((size_t)bb * M + m) * N + n0;
        out[i0]      = acc0[j] + bv + xres[i0];
        out[i0 + 16] = acc1[j] + bv + xres[i0 + 16];
    }
}

// ---------------------------------------------------------------------------
// Attention: one block per (head-batch, 16-row t-tile). T=1024, ch=64.
//   phase 1: S[16][1024] = (q*s)^T (k*s) via WMMA  -> LDS (f32)
//   phase 2: row softmax (shfl reductions, v_exp)  -> P^T[1024][16] bf16 LDS
//   phase 3: a[c][t] = V x P^T via WMMA, K split across wave pairs
// LDS: 64KB S + 32KB P + 2KB q = ~98KB -> 3 workgroups / WGP.
// ---------------------------------------------------------------------------
__global__ void __launch_bounds__(256) attn_kernel(const __bf16* __restrict__ qkv,
                                                   __bf16* __restrict__ aout) {
    int bh = blockIdx.y;               // 0..63
    int b = bh >> 3, h = bh & 7;
    int tb = blockIdx.x * 16;          // t-tile base

    const __bf16* qp = qkv + ((size_t)b * (3 * C_) + h * (3 * CH_)) * T_;
    const __bf16* kp = qp + (size_t)CH_ * T_;
    const __bf16* vp = qp + (size_t)2 * CH_ * T_;

    __shared__ float  sS[16 * 1024];                // 64 KB score strip / reduce scratch
    __shared__ __align__(16) __bf16 sP[1024 * 16];  // 32 KB P^T (s-major)
    __shared__ __bf16 sQ[16 * 64];                  // 2 KB scaled q^T

    int tid = threadIdx.x;
    int w = tid >> 5, l = tid & 31, g = l >> 4, ln = l & 15;

    // stage q^T, fold in scale^2 = 1/sqrt(ch) = 0.125 (exact power of two)
    for (int i = tid; i < 16 * 64; i += 256) {
        int t = i & 15, c = i >> 4;
        sQ[t * 64 + c] = (__bf16)((float)qp[(size_t)c * T_ + tb + t] * 0.125f);
    }
    __syncthreads();

    // ---- phase 1: S = q^T k, wave w owns s in [w*128, w*128+128) ----
    v16bf aq0, aq1;
    {
        const __bf16* ap = &sQ[ln * 64];
        #pragma unroll
        for (int e = 0; e < 16; ++e) {
            int pat = (e < 8 ? e : e + 8) + g * 8;
            aq0[e] = ap[pat];        // K = c in [0,32)
            aq1[e] = ap[32 + pat];   // K = c in [32,64)
        }
    }
    int kl = ln + g * 16;
    for (int nt = 0; nt < 8; ++nt) {
        int s0 = w * 128 + nt * 16;
        v8f acc = {};
        v16bf bk;
        const __bf16* bp0 = &kp[s0 + ln];
        #pragma unroll
        for (int e = 0; e < 16; ++e) bk[e] = bp0[(size_t)((e < 8 ? e : e + 8) + g * 8) * T_];
        acc = __builtin_amdgcn_wmma_f32_16x16x32_bf16(false, aq0, false, bk, (short)0, acc, false, false);
        const __bf16* bp1 = &kp[(size_t)32 * T_ + s0 + ln];
        #pragma unroll
        for (int e = 0; e < 16; ++e) bk[e] = bp1[(size_t)((e < 8 ? e : e + 8) + g * 8) * T_];
        acc = __builtin_amdgcn_wmma_f32_16x16x32_bf16(false, aq1, false, bk, (short)0, acc, false, false);
        #pragma unroll
        for (int j = 0; j < 8; ++j)
            sS[(g * 8 + j) * 1024 + s0 + ln] = acc[j];
    }
    __syncthreads();

    // ---- phase 2: softmax over s for rows w*2, w*2+1 ----
    #pragma unroll
    for (int rr = 0; rr < 2; ++rr) {
        int r = w * 2 + rr;
        float vals[32];
        float mx = -3.0e38f;
        #pragma unroll
        for (int i = 0; i < 32; ++i) {
            vals[i] = sS[r * 1024 + l + i * 32];
            mx = fmaxf(mx, vals[i]);
        }
        #pragma unroll
        for (int off = 16; off > 0; off >>= 1) mx = fmaxf(mx, __shfl_xor(mx, off, 32));
        float sum = 0.f;
        #pragma unroll
        for (int i = 0; i < 32; ++i) { vals[i] = __expf(vals[i] - mx); sum += vals[i]; }
        #pragma unroll
        for (int off = 16; off > 0; off >>= 1) sum += __shfl_xor(sum, off, 32);
        float inv = 1.f / sum;
        #pragma unroll
        for (int i = 0; i < 32; ++i)
            sP[(l + i * 32) * 16 + r] = (__bf16)(vals[i] * inv);  // transposed P
    }
    __syncthreads();

    // ---- phase 3: a = V * P^T. wave w: c-tile (w&3), K-half (w>>2) ----
    int mt = w & 3, kh = w >> 2;
    v8f acc = {};
    for (int ck = 0; ck < 16; ++ck) {
        int s0 = kh * 512 + ck * 32;
        v16bf av, bpv;
        const __bf16* ap = &vp[(size_t)(mt * 16 + ln) * T_ + s0];
        #pragma unroll
        for (int e = 0; e < 16; ++e) av[e] = ap[(e < 8 ? e : e + 8) + g * 8];
        const __bf16* bp = &sP[s0 * 16 + ln];
        #pragma unroll
        for (int e = 0; e < 16; ++e) bpv[e] = bp[((e < 8 ? e : e + 8) + g * 8) * 16];
        acc = __builtin_amdgcn_wmma_f32_16x16x32_bf16(false, av, false, bpv, (short)0, acc, false, false);
    }
    // combine the two K-halves through LDS (sS is free now)
    if (kh == 1) {
        float* red = &sS[mt * 256];
        #pragma unroll
        for (int j = 0; j < 8; ++j) red[l * 8 + j] = acc[j];
    }
    __syncthreads();
    if (kh == 0) {
        const float* red = &sS[mt * 256];
        #pragma unroll
        for (int j = 0; j < 8; ++j) {
            float vsum = acc[j] + red[l * 8 + j];
            int cg = h * 64 + mt * 16 + g * 8 + j;
            aout[((size_t)b * C_ + cg) * T_ + tb + ln] = (__bf16)vsum;
        }
    }
}

// ---------------------------------------------------------------------------
// launcher
// ---------------------------------------------------------------------------
extern "C" void kernel_launch(void* const* d_in, const int* in_sizes, int n_in,
                              void* d_out, int out_size, void* d_ws, size_t ws_size,
                              hipStream_t stream) {
    const float* x      = (const float*)d_in[0];
    const float* norm_w = (const float*)d_in[1];
    const float* norm_b = (const float*)d_in[2];
    const float* qkv_w  = (const float*)d_in[3];
    const float* qkv_b  = (const float*)d_in[4];
    const float* proj_w = (const float*)d_in[5];
    const float* proj_b = (const float*)d_in[6];
    float* out = (float*)d_out;

    char* ws = (char*)d_ws;
    size_t off = 0;
    auto alloc = [&](size_t bytes) -> void* {
        void* p = ws + off;
        off = (off + bytes + 255) & ~(size_t)255;
        return p;
    };
    __bf16* xn_bf    = (__bf16*)alloc((size_t)B_ * C_ * T_ * 2);        // 8 MB
    __bf16* qkvw_bf  = (__bf16*)alloc((size_t)3 * C_ * C_ * 2);         // 1.5 MB
    __bf16* projw_bf = (__bf16*)alloc((size_t)C_ * C_ * 2);             // 0.5 MB
    __bf16* qkv_bf   = (__bf16*)alloc((size_t)B_ * 3 * C_ * T_ * 2);    // 24 MB
    __bf16* a_bf     = (__bf16*)alloc((size_t)B_ * C_ * T_ * 2);        // 8 MB

    cvt_bf16_kernel<<<(3 * C_ * C_ + 255) / 256, 256, 0, stream>>>(qkv_w, qkvw_bf, 3 * C_ * C_);
    cvt_bf16_kernel<<<(C_ * C_ + 255) / 256, 256, 0, stream>>>(proj_w, projw_bf, C_ * C_);
    groupnorm_kernel<<<B_ * 32, 256, 0, stream>>>(x, norm_w, norm_b, xn_bf);
    qkv_gemm_kernel<<<dim3(T_ / 64, (3 * C_) / 64, B_), 256, 0, stream>>>(qkvw_bf, xn_bf, qkv_b, qkv_bf);
    attn_kernel<<<dim3(T_ / 16, B_ * NH_), 256, 0, stream>>>(qkv_bf, a_bf);
    proj_gemm_kernel<<<dim3(T_ / 64, C_ / 64, B_), 256, 0, stream>>>(projw_bf, a_bf, proj_b, x, out);
}
